// AnomalyAttention_13726715478514
// MI455X (gfx1250) — matrix operand order repaired and checked by reading, not compiled, not run on hardware
//
#include <hip/hip_runtime.h>
#include <hip/hip_bf16.h>
#include <cmath>

// ---------------- CDNA5 WMMA / TDM types ----------------
typedef _Float16     h16   __attribute__((ext_vector_type(16)));
typedef _Float16     h8    __attribute__((ext_vector_type(8)));
typedef float        f8    __attribute__((ext_vector_type(8)));
typedef unsigned int u32x4 __attribute__((ext_vector_type(4)));
typedef int          i32x4 __attribute__((ext_vector_type(4)));
typedef int          i32x8 __attribute__((ext_vector_type(8)));

#define B_  8
#define L_  1024
#define H_  8
#define E_  64
#define NT  (L_ / 16)   // 64 column tiles

// ---- small helpers ----
__device__ inline void ld8(const float* p, float* d) {
    float4 a = ((const float4*)p)[0];
    float4 b = ((const float4*)p)[1];
    d[0] = a.x; d[1] = a.y; d[2] = a.z; d[3] = a.w;
    d[4] = b.x; d[5] = b.y; d[6] = b.z; d[7] = b.w;
}

// 16 contiguous f32 -> h16 (K B-fragment: lane n holds col n, halves t -> k = khi + t)
__device__ inline h16 load_h16(const float* p, float s) {
    float t[16];
    ld8(p, t); ld8(p + 8, t + 8);
    h16 r;
#pragma unroll
    for (int i = 0; i < 16; ++i) r[i] = (_Float16)(t[i] * s);
    return r;
}

// Q A-fragment: lane m = lane&15, hi = lane>>4; halves t -> k = koff+hi*16+t
// (contiguous order, matching the B fragments)
__device__ inline h16 load_a_qfrag(const float* qrow, int koff, int hi, float s) {
    float t0[8], t1[8];
    ld8(qrow + koff + hi * 16, t0);
    ld8(qrow + koff + hi * 16 + 8, t1);
    h16 r;
#pragma unroll
    for (int t = 0; t < 8; ++t) {
        r[t]     = (_Float16)(t0[t] * s);
        r[t + 8] = (_Float16)(t1[t] * s);
    }
    return r;
}

__device__ inline h16 cat8(h8 lo, h8 hi8) {
    h16 r;
#pragma unroll
    for (int t = 0; t < 8; ++t) { r[t] = lo[t]; r[t + 8] = hi8[t]; }
    return r;
}

__device__ inline float gmax16(float v) {   // reduce over 16-lane half (xor masks stay in-half)
    v = fmaxf(v, __shfl_xor(v, 1));
    v = fmaxf(v, __shfl_xor(v, 2));
    v = fmaxf(v, __shfl_xor(v, 4));
    v = fmaxf(v, __shfl_xor(v, 8));
    return v;
}
__device__ inline float gsum16(float v) {
    v += __shfl_xor(v, 1);
    v += __shfl_xor(v, 2);
    v += __shfl_xor(v, 4);
    v += __shfl_xor(v, 8);
    return v;
}

// one 16x16 score tile: scores[M][N] = sum_k Qs[m][k] * Key[j*16+n][k] (Q pre-scaled)
__device__ inline f8 qk_tile(const h16* aQ, const float* Kg, int b, int h, int j,
                             int n, int khi, f8 c) {
    const float* krow = Kg + ((size_t)((b * L_ + j * 16 + n) * H_ + h)) * E_;
#pragma unroll
    for (int f = 0; f < 2; ++f) {
        h16 bK = load_h16(krow + f * 32 + khi, 1.0f);
        c = __builtin_amdgcn_wmma_f32_16x16x32_f16(false, aQ[f], false, bK,
                                                   (short)0, c, false, false);
    }
    return c;
}

// ---- Tensor Data Mover: DMA a 32-row x 64-col f32 tile (row stride 512 elems)
// from global memory into LDS at lds_off (per-wave uniform). Tracked by TENSORcnt.
__device__ inline void tdm_load_v_tile(const float* gtile, unsigned lds_off) {
    unsigned long long ga = (unsigned long long)(size_t)gtile;
    u32x4 g0;
    g0[0] = 1u;                                        // count=1 (valid user descriptor)
    g0[1] = lds_off;                                   // lds_addr (bytes)
    g0[2] = (unsigned)(ga & 0xFFFFFFFFull);            // global_addr[31:0]
    g0[3] = (unsigned)((ga >> 32) & 0x1FFFFFFull)      // global_addr[56:32]
          | (2u << 30);                                // type = 2 ("image")
    i32x8 g1;
    g1[0] = (int)(2u << 16);                           // data_size = 2 -> 4 bytes
    g1[1] = (int)(64u << 16);                          // tensor_dim0[15:0] = 64
    g1[2] = (int)(1024u << 16);                        // tensor_dim1[15:0] = 1024
    g1[3] = (int)(64u << 16);                          // tile_dim0 = 64
    g1[4] = (int)32u;                                  // tile_dim1 = 32, tile_dim2 = 0
    g1[5] = (int)512u;                                 // tensor_dim0_stride = H*E = 512
    g1[6] = 0;                                         // stride0 hi / dim1_stride lo
    g1[7] = 0;                                         // dim1_stride hi
    i32x4 z4; z4[0] = 0; z4[1] = 0; z4[2] = 0; z4[3] = 0;
#if __clang_major__ >= 23
    i32x8 z8; z8[0]=0; z8[1]=0; z8[2]=0; z8[3]=0; z8[4]=0; z8[5]=0; z8[6]=0; z8[7]=0;
    __builtin_amdgcn_tensor_load_to_lds(g0, g1, z4, z4, z8, 0);
#else
    __builtin_amdgcn_tensor_load_to_lds(g0, g1, z4, z4, 0);
#endif
}

// ================= fused causal attention: scores -> softmax -> probs + O =================
__launch_bounds__(128)
__global__ void anomaly_attn_kernel(const float* __restrict__ Q,
                                    const float* __restrict__ K,
                                    const float* __restrict__ V,
                                    float* __restrict__ Out,     // [B,L,H,E]
                                    float* __restrict__ Ser) {   // [B,H,L,L]
    __shared__ __align__(16) _Float16 Pbuf[4][16][32];   // per-wave P tile pair (A-source)
    __shared__ __align__(16) float    Vf32[4][32][64];   // per-wave TDM-staged V tile pair

    const int lane = threadIdx.x & 31;
    const int w    = threadIdx.x >> 5;
    const int hi   = lane >> 4;          // which 16-lane half
    const int n    = lane & 15;          // col index (B/C layouts) == row m (A layout)
    const int khi  = hi * 16;            // B-fragment K offset for this half
    const int bh   = blockIdx.y;
    const int b    = bh >> 3;
    const int h    = bh & 7;
    const int i    = blockIdx.x * 4 + w; // query row tile (16 rows)

    const float scale = 0.125f;          // 1/sqrt(E), folded into Q before f16 convert

    // ---- Q A-fragments (row m = i*16 + n for this lane) ----
    const float* qrow = Q + ((size_t)((b * L_ + i * 16 + n) * H_ + h)) * E_;
    h16 aQ[2];
    aQ[0] = load_a_qfrag(qrow, 0,  hi, scale);
    aQ[1] = load_a_qfrag(qrow, 32, hi, scale);

    // ---- Pass A: online row max / sum over causal range ----
    float m[8], l[8];
#pragma unroll
    for (int v = 0; v < 8; ++v) { m[v] = -INFINITY; l[v] = 0.0f; }

    for (int j = 0; j <= i; ++j) {
        // prefetch next tile's K rows (L2-resident; global_prefetch_b8)
        {
            const int jn = (j < i) ? j + 1 : j;
            __builtin_prefetch(K + ((size_t)((b * L_ + jn * 16 + n) * H_ + h)) * E_ + khi, 0, 0);
        }
        f8 c;
#pragma unroll
        for (int v = 0; v < 8; ++v) c[v] = 0.0f;
        c = qk_tile(aQ, K, b, h, j, n, khi, c);
#pragma unroll
        for (int v = 0; v < 8; ++v) {
            const int M = v + hi * 8;                 // row within tile
            float val = c[v];
            if (j == i && n > M) val = -INFINITY;     // causal mask on diagonal tile
            float nm = fmaxf(m[v], gmax16(val));
            float e  = __expf(val - nm);
            float gs = gsum16(e);
            l[v] = l[v] * __expf(m[v] - nm) + gs;
            m[v] = nm;
        }
    }

    float rl[8];
#pragma unroll
    for (int v = 0; v < 8; ++v) rl[v] = 1.0f / l[v];

    // ---- Pass B: write probs once, accumulate O = P*V with WMMA ----
    f8 acc[4];
#pragma unroll
    for (int q = 0; q < 4; ++q)
#pragma unroll
        for (int v = 0; v < 8; ++v) acc[q][v] = 0.0f;

    const unsigned vlds = (unsigned)(size_t)&Vf32[w][0][0];   // LDS byte offset (aperture low bits)

    for (int j0 = 0; j0 <= i; j0 += 2) {
        // kick off the V tile-pair DMA early: WAR-fence previous iteration's LDS reads,
        // then let the TDM run while we compute the P tiles below.
        asm volatile("s_wait_dscnt 0x0" ::: "memory");
        tdm_load_v_tile(V + ((size_t)((b * L_ + j0 * 16) * H_ + h)) * E_, vlds);

#pragma unroll
        for (int half = 0; half < 2; ++half) {
            const int jj = j0 + half;
            if (jj <= i) {
                f8 c;
#pragma unroll
                for (int v = 0; v < 8; ++v) c[v] = 0.0f;
                c = qk_tile(aQ, K, b, h, jj, n, khi, c);
#pragma unroll
                for (int v = 0; v < 8; ++v) {
                    const int M = v + hi * 8;
                    float val = c[v];
                    if (jj == i && n > M) val = -INFINITY;
                    float p = __expf(val - m[v]) * rl[v];
                    Ser[((size_t)bh * L_ + i * 16 + M) * L_ + jj * 16 + n] = p;
                    Pbuf[w][M][half * 16 + n] = (_Float16)p;
                }
            } else {
#pragma unroll
                for (int v = 0; v < 8; ++v)
                    Pbuf[w][v + hi * 8][16 + n] = (_Float16)0.0f;
            }
        }

        // A-fragment of P (16x32): lane m = n; halves t -> k = hi*16+t
        h16 aP = cat8(*(const h8*)&Pbuf[w][n][hi * 16],
                      *(const h8*)&Pbuf[w][n][hi * 16 + 8]);

        // wait for the TDM tile, then build V B-fragments from LDS:
        // B[k][nn] = V[j0*16+k][q*16+nn] = Vf32[khi+t][q*16+nn]
        __builtin_amdgcn_s_wait_tensorcnt(0);
#pragma unroll
        for (int q = 0; q < 4; ++q) {
            h16 bV;
#pragma unroll
            for (int t = 0; t < 16; ++t)
                bV[t] = (_Float16)Vf32[w][khi + t][q * 16 + n];
            acc[q] = __builtin_amdgcn_wmma_f32_16x16x32_f16(false, aP, false, bV,
                                                            (short)0, acc[q], false, false);
        }
    }

    // zero-fill strictly-upper (masked) prob tiles — d_out is poisoned by harness
    for (int j = i + 1; j < NT; ++j) {
#pragma unroll
        for (int v = 0; v < 8; ++v)
            Ser[((size_t)bh * L_ + i * 16 + v + hi * 8) * L_ + j * 16 + n] = 0.0f;
    }

    // write O tile: D layout -> row M = v + hi*8, col e = q*16 + n
#pragma unroll
    for (int q = 0; q < 4; ++q)
#pragma unroll
        for (int v = 0; v < 8; ++v)
            Out[((size_t)((b * L_ + i * 16 + v + hi * 8) * H_ + h)) * E_ + q * 16 + n] = acc[q][v];
}

// ================= Gaussian prior (inv_norm cancels in the normalization) =================
__launch_bounds__(128)
__global__ void prior_kernel(const float* __restrict__ Sig,   // [B,L,H]
                             float* __restrict__ Prior) {     // [B,H,L,L]
    const int lane = threadIdx.x & 31;
    const int wid  = blockIdx.x * 4 + (threadIdx.x >> 5);     // row id = (b*H+h)*L + l
    const int b    = wid >> 13;            // / (H*L) = / 8192
    const int rem  = wid & 8191;
    const int h    = rem >> 10;            // / L
    const int l    = rem & 1023;

    const float sg = Sig[((size_t)b * L_ + l) * H_ + h];
    const float c  = -1.0f / (2.0f * sg * sg);

    float ev[32];
    float sum = 0.0f;
#pragma unroll
    for (int k = 0; k < 32; ++k) {
        const int s = k * 32 + lane;
        const float d = (float)(l - s);
        const float e = __expf(d * d * c);
        ev[k] = e;
        sum  += e;
    }
    sum += __shfl_xor(sum, 1);
    sum += __shfl_xor(sum, 2);
    sum += __shfl_xor(sum, 4);
    sum += __shfl_xor(sum, 8);
    sum += __shfl_xor(sum, 16);
    const float inv = 1.0f / sum;

    float* row = Prior + (size_t)wid * L_;
#pragma unroll
    for (int k = 0; k < 32; ++k)
        row[k * 32 + lane] = ev[k] * inv;
}

// =================================== launch ===================================
extern "C" void kernel_launch(void* const* d_in, const int* in_sizes, int n_in,
                              void* d_out, int out_size, void* d_ws, size_t ws_size,
                              hipStream_t stream) {
    (void)in_sizes; (void)n_in; (void)out_size; (void)d_ws; (void)ws_size;

    const float* Q   = (const float*)d_in[0];
    const float* K   = (const float*)d_in[1];
    const float* V   = (const float*)d_in[2];
    const float* Sig = (const float*)d_in[3];

    float* out   = (float*)d_out;
    float* Out0  = out;                                          // [B,L,H,E]
    float* Ser   = out + (size_t)B_ * L_ * H_ * E_;              // [B,H,L,L]
    float* Prior = Ser + (size_t)B_ * H_ * L_ * L_;              // [B,H,L,L]

    anomaly_attn_kernel<<<dim3(L_ / 64, B_ * H_), 128, 0, stream>>>(Q, K, V, Out0, Ser);
    prior_kernel<<<dim3((B_ * H_ * L_) / 4), 128, 0, stream>>>(Sig, Prior);
}
